// Head_2637109920377
// MI455X (gfx1250) — matrix operand order repaired and checked by reading, not compiled, not run on hardware
//
#include <hip/hip_runtime.h>

// ---------------------------------------------------------------------------
// Single-head causal attention for MI455X (gfx1250), wave32 + WMMA f16.
//   x  : [B=8, T=2048, C=1024] f32
//   Wq/Wk/Wv : [1024, 64] f32
//   out: [8, 2048, 64] f32
// Pipeline:
//   phase 0: W* -> f16, transposed [3][64][1024]     (B-frags contiguous)
//   phase 1: fused QKV, 1 wave per 16-row tile, 12 accumulators,
//            all-loads-then-all-WMMAs + x prefetch software pipeline
//            q,k -> f16 [B*T][64];  v -> f16 transposed [B][64][T]
//   phase 2: flash attention, 1 wave per (b, 16-query tile), 32 keys/iter,
//            K/V fragment loads hoisted above softmax to overlap latency
// ---------------------------------------------------------------------------

typedef __attribute__((ext_vector_type(16))) _Float16 v16h;
typedef __attribute__((ext_vector_type(8)))  float    v8f;

#define WMMA_F16(A, B, Cacc) \
  __builtin_amdgcn_wmma_f32_16x16x32_f16(false, (A), false, (B), (short)0, (Cacc), false, false)

static constexpr int BATCH = 8;
static constexpr int TSEQ  = 2048;
static constexpr int CDIM  = 1024;
static constexpr int HDIM  = 64;

union AFrag { v16h v; _Float16 h[16]; };

static __device__ __forceinline__ v8f v8f_zero() {
  v8f z = {0.f, 0.f, 0.f, 0.f, 0.f, 0.f, 0.f, 0.f};
  return z;
}

// ---------------------------------------------------------------------------
// Phase 0: convert & transpose weights: wt[(mat*64+n)*1024 + k] = W[mat][k][n]
// ---------------------------------------------------------------------------
__global__ void wt_convert_kernel(const float* __restrict__ Wq,
                                  const float* __restrict__ Wk,
                                  const float* __restrict__ Wv,
                                  _Float16* __restrict__ wt) {
  int idx = blockIdx.x * 256 + threadIdx.x;          // 0 .. 3*64*1024-1
  int mat = idx >> 16;                               // 65536 elements per matrix
  int rem = idx & 65535;
  int n   = rem >> 10;                               // 0..63
  int k   = rem & 1023;                              // 0..1023
  const float* W = (mat == 0) ? Wq : (mat == 1) ? Wk : Wv;
  wt[idx] = (_Float16)W[k * HDIM + n];               // idx == (mat*64+n)*1024 + k
}

// ---------------------------------------------------------------------------
// Phase 1: fused QKV projection. grid.x = B*T/16 row tiles, 32 threads (1 wave).
// ---------------------------------------------------------------------------
__global__ __launch_bounds__(32)
void qkv_proj_kernel(const float* __restrict__ x,
                     const _Float16* __restrict__ wt,
                     _Float16* __restrict__ qh,
                     _Float16* __restrict__ kh,
                     _Float16* __restrict__ vth) {
  const int rt   = blockIdx.x;                 // 16-row tile over B*T
  const int lane = threadIdx.x & 31;
  const int lh   = lane >> 4;                  // half-wave: 0/1
  const int lr   = lane & 15;                  // row (A) / col (B,C,D)

  v8f acc[12];
  #pragma unroll
  for (int i = 0; i < 12; ++i) acc[i] = v8f_zero();

  const float* xrow = x + ((size_t)rt * 16 + lr) * CDIM;

  // A-fragment (16x32 f16): lane lr = row, K runs [kc+16lh,+16) contiguous
  auto load_a = [&](int kc) {
    AFrag a;
    const float* p0 = xrow + kc + 16 * lh;
    const float* p1 = xrow + kc + 16 * lh + 8;
    #pragma unroll
    for (int j = 0; j < 8; ++j) {
      a.h[j]     = (_Float16)p0[j];
      a.h[8 + j] = (_Float16)p1[j];
    }
    return a;
  };

  AFrag a = load_a(0);

  for (int kc = 0; kc < CDIM; kc += 32) {
    // 1) issue all 12 B-fragment loads as one clause (each 32 B contiguous)
    v16h bf[12];
    #pragma unroll
    for (int m = 0; m < 3; ++m)
      #pragma unroll
      for (int c = 0; c < 4; ++c)
        bf[m * 4 + c] = *(const v16h*)(wt + ((size_t)(m * HDIM + c * 16 + lr)) * CDIM
                                          + kc + 16 * lh);
    // 2) prefetch next x chunk (wraps harmlessly on the last iteration)
    AFrag an = load_a((kc + 32) & (CDIM - 1));
    // 3) 12 back-to-back WMMAs on the current A-fragment
    #pragma unroll
    for (int i = 0; i < 12; ++i)
      acc[i] = WMMA_F16(a.v, bf[i], acc[i]);
    a = an;
  }

  // C/D layout: col = lr, row = r + 8*lh
  #pragma unroll
  for (int c = 0; c < 4; ++c) {
    #pragma unroll
    for (int r = 0; r < 8; ++r) {
      int row = rt * 16 + r + 8 * lh;          // global row over B*T
      int col = c * 16 + lr;
      qh[(size_t)row * HDIM + col] = (_Float16)acc[0 * 4 + c][r];
      kh[(size_t)row * HDIM + col] = (_Float16)acc[1 * 4 + c][r];
      int b = row >> 11, t = row & (TSEQ - 1);
      vth[((size_t)b * HDIM + col) * TSEQ + t] = (_Float16)acc[2 * 4 + c][r];
    }
  }
}

// ---------------------------------------------------------------------------
// Phase 2: flash attention. grid = (T/16, B), 32 threads (1 wave).
// ---------------------------------------------------------------------------
__global__ __launch_bounds__(32)
void attn_kernel(const _Float16* __restrict__ qh,
                 const _Float16* __restrict__ kh,
                 const _Float16* __restrict__ vth,
                 float* __restrict__ out) {
  const float scale = 0.03125f;                // C**-0.5 = 1/32 (C = 1024)
  const int qt   = blockIdx.x;                 // query tile 0..127
  const int b    = blockIdx.y;                 // batch
  const int lane = threadIdx.x & 31;
  const int lh   = lane >> 4;
  const int lr   = lane & 15;

  __shared__ alignas(32) _Float16 Plds[16][32];

  const _Float16* qb = qh  + (size_t)b * TSEQ * HDIM;
  const _Float16* kb = kh  + (size_t)b * TSEQ * HDIM;
  const _Float16* vb = vth + (size_t)b * HDIM * TSEQ;

  // Q A-fragments (16x64 -> two K=32 fragments), held in registers all loop
  AFrag qa[2];
  {
    const _Float16* qrow = qb + ((size_t)qt * 16 + lr) * HDIM;
    #pragma unroll
    for (int kc2 = 0; kc2 < 2; ++kc2) {
      int kc = kc2 * 32;
      #pragma unroll
      for (int j = 0; j < 8; ++j) {
        qa[kc2].h[j]     = qrow[kc + 16 * lh + j];
        qa[kc2].h[8 + j] = qrow[kc + 16 * lh + 8 + j];
      }
    }
  }

  float m_run[8], l_run[8];
  v8f O[4];
  #pragma unroll
  for (int r = 0; r < 8; ++r) { m_run[r] = -1e30f; l_run[r] = 0.f; }
  #pragma unroll
  for (int c = 0; c < 4; ++c) O[c] = v8f_zero();

  const int qlast = qt * 16 + 15;

  for (int kbase = 0; kbase <= qlast; kbase += 32) {
    // ---- hoist ALL global fragment loads for this iteration (one clause) ----
    v16h kf0[2], kf1[2], vf[4];
    #pragma unroll
    for (int s = 0; s < 2; ++s) {
      const _Float16* krow = kb + ((size_t)(kbase + s * 16 + lr)) * HDIM;
      kf0[s] = *(const v16h*)(krow + 16 * lh);         // K-dims 0..31
      kf1[s] = *(const v16h*)(krow + 32 + 16 * lh);    // K-dims 32..63
    }
    #pragma unroll
    for (int c = 0; c < 4; ++c)
      vf[c] = *(const v16h*)(vb + ((size_t)(c * 16 + lr)) * TSEQ
                                + kbase + 16 * lh);

    // ---- scores: S[s] = Q(16x64) x K^T(64x16) ----
    v8f S[2];
    S[0] = v8f_zero(); S[1] = v8f_zero();
    #pragma unroll
    for (int s = 0; s < 2; ++s) {
      S[s] = WMMA_F16(qa[0].v, kf0[s], S[s]);
      S[s] = WMMA_F16(qa[1].v, kf1[s], S[s]);
    }

    // ---- online softmax (f32), per row M = r + 8*lh across 16 lanes ----
    // (VALU work here overlaps the outstanding vf[] loads)
    float corr[8];
    #pragma unroll
    for (int r = 0; r < 8; ++r) {
      int   qrow = qt * 16 + r + 8 * lh;
      float s0 = S[0][r] * scale;
      float s1 = S[1][r] * scale;
      if (kbase + lr > qrow)      s0 = -1e30f;         // causal mask
      if (kbase + 16 + lr > qrow) s1 = -1e30f;
      float mx = fmaxf(s0, s1);
      #pragma unroll
      for (int off = 1; off < 16; off <<= 1)
        mx = fmaxf(mx, __shfl_xor(mx, off, 32));
      float mnew = fmaxf(m_run[r], mx);
      float p0 = __expf(s0 - mnew);
      float p1 = __expf(s1 - mnew);
      float cf = __expf(m_run[r] - mnew);
      float rs = p0 + p1;
      #pragma unroll
      for (int off = 1; off < 16; off <<= 1)
        rs += __shfl_xor(rs, off, 32);
      l_run[r] = l_run[r] * cf + rs;
      m_run[r] = mnew;
      corr[r]  = cf;
      Plds[r + 8 * lh][lr]      = (_Float16)p0;
      Plds[r + 8 * lh][16 + lr] = (_Float16)p1;
    }
    #pragma unroll
    for (int c = 0; c < 4; ++c)
      #pragma unroll
      for (int r = 0; r < 8; ++r) O[c][r] *= corr[r];

    __syncthreads();   // single-wave WG: orders LDS write -> read

    // ---- O += P(16x32) x V(32x64) ----
    AFrag pa;
    #pragma unroll
    for (int j = 0; j < 8; ++j) {
      pa.h[j]     = Plds[lr][16 * lh + j];
      pa.h[8 + j] = Plds[lr][16 * lh + 8 + j];
    }
    #pragma unroll
    for (int c = 0; c < 4; ++c)
      O[c] = WMMA_F16(pa.v, vf[c], O[c]);

    __syncthreads();
  }

  // ---- epilogue: out = O / l ----
  #pragma unroll
  for (int r = 0; r < 8; ++r) {
    float inv = 1.0f / l_run[r];
    int row = qt * 16 + r + 8 * lh;
    #pragma unroll
    for (int c = 0; c < 4; ++c)
      out[((size_t)b * TSEQ + row) * HDIM + c * 16 + lr] = O[c][r] * inv;
  }
}

// ---------------------------------------------------------------------------
extern "C" void kernel_launch(void* const* d_in, const int* in_sizes, int n_in,
                              void* d_out, int out_size, void* d_ws, size_t ws_size,
                              hipStream_t stream) {
  (void)in_sizes; (void)n_in; (void)out_size; (void)ws_size;
  const float* x  = (const float*)d_in[0];
  const float* Wq = (const float*)d_in[1];
  const float* Wk = (const float*)d_in[2];
  const float* Wv = (const float*)d_in[3];
  float* out = (float*)d_out;

  // workspace layout (bytes):
  //   wt : 3*64*1024*2        = 393216
  //   qh : 8*2048*64*2        = 2097152
  //   kh : 2097152
  //   vth: 2097152            total ~6.7 MB
  char* ws = (char*)d_ws;
  _Float16* wt  = (_Float16*)(ws);
  _Float16* qh  = (_Float16*)(ws + 393216);
  _Float16* kh  = (_Float16*)(ws + 393216 + 2097152);
  _Float16* vth = (_Float16*)(ws + 393216 + 2u * 2097152);

  wt_convert_kernel<<<768, 256, 0, stream>>>(Wq, Wk, Wv, wt);
  qkv_proj_kernel<<<dim3(BATCH * TSEQ / 16), 32, 0, stream>>>(x, wt, qh, kh, vth);
  attn_kernel<<<dim3(TSEQ / 16, BATCH), 32, 0, stream>>>(qh, kh, vth, out);
}
